// LSTM_25958782337436
// MI455X (gfx1250) — matrix operand order repaired and checked by reading, not compiled, not run on hardware
//
#include <hip/hip_runtime.h>
#include <stdint.h>

typedef __attribute__((ext_vector_type(16))) __bf16 v16bf;
typedef __attribute__((ext_vector_type(8)))  float  v8f;

#define H  1024
#define Bn 256
#define T  256
#define Cc 10
#define KB (H / 32)    // 32 k-blocks of depth 32
#define MT (H / 16)    // 64 m tiles
#define FRAG 512       // ushorts per A fragment (32 lanes * 16 elems)

__device__ __forceinline__ unsigned short f2bf(float f) {
  union { float f; unsigned u; } v; v.f = f;
  unsigned r = v.u + 0x7FFFu + ((v.u >> 16) & 1u);   // round-to-nearest-even
  return (unsigned short)(r >> 16);
}
__device__ __forceinline__ float sigm(float x) {
  return 1.0f / (1.0f + __expf(-x));
}
__device__ __forceinline__ float tanh_fast(float x) {
  x = fminf(fmaxf(x, -15.0f), 15.0f);
  float e = __expf(2.0f * x);
  return (e - 1.0f) / (e + 1.0f);
}

// ---------------------------------------------------------------------------
// Pack the 4 recurrent weight matrices (fp32, row-major [H][H]) into bf16
// WMMA A-fragment order: [gate][m_tile][k_block][lane][16 elems].
// A 16x32 layout (contiguous k, matching B): lane L (<16) row=L, elems -> K {0..15};
// lane L+16 row=L, elems -> K {16..31}.
// ---------------------------------------------------------------------------
__global__ void pack_w(const float* __restrict__ Wgh, const float* __restrict__ Wih,
                       const float* __restrict__ Wfh, const float* __restrict__ Woh,
                       unsigned short* __restrict__ Wp) {
  int idx  = blockIdx.x * blockDim.x + threadIdx.x;   // 0 .. 4*H*H-1
  int e    = idx & 15;
  int lane = (idx >> 4) & 31;
  int kb   = (idx >> 9) & 31;
  int tm   = (idx >> 14) & 63;
  int g    = idx >> 20;
  int K    = kb * 32 + e + ((lane >> 4) << 4);
  int row  = tm * 16 + (lane & 15);
  const float* W = (g == 0) ? Wgh : (g == 1) ? Wih : (g == 2) ? Wfh : Woh;
  Wp[idx] = f2bf(W[row * H + K]);
}

__global__ void init_state(float* __restrict__ cst, unsigned short* __restrict__ h0) {
  int idx = blockIdx.x * blockDim.x + threadIdx.x;
  if (idx < H * Bn) { cst[idx] = 0.0f; h0[idx] = 0; }
}

// ---------------------------------------------------------------------------
// One LSTM time step. Per wave: 16(M) x 32(N) output tile for all 4 gates.
// Grid = 64 blocks x 256 threads = 512 waves = (64 m-tiles) x (8 n-groups).
// h is bf16 stored [batch][H] so each B-fragment lane load is contiguous.
// ---------------------------------------------------------------------------
__global__ __launch_bounds__(256)
void lstm_step(const unsigned short* __restrict__ Wp,
               const unsigned short* __restrict__ hsrc,
               unsigned short* __restrict__ hdst,
               float* __restrict__ hf32,
               float* __restrict__ cst,
               const float* __restrict__ x,
               const float* __restrict__ Wgx, const float* __restrict__ Wix,
               const float* __restrict__ Wfx, const float* __restrict__ Wox,
               const float* __restrict__ bgp, const float* __restrict__ bip,
               const float* __restrict__ bfp, const float* __restrict__ bop,
               int t) {
  int wid  = (blockIdx.x * blockDim.x + threadIdx.x) >> 5;
  int lane = threadIdx.x & 31;
  int mt   = wid & 63;            // m tile
  int ng   = wid >> 6;            // n group (0..7), 32 columns each
  int m0   = mt * 16;
  int n0   = ng * 32;

  v8f aG0 = {}, aG1 = {}, aI0 = {}, aI1 = {};
  v8f aF0 = {}, aF1 = {}, aO0 = {}, aO1 = {};

  int klo = lane & 16;            // B layout: high half-wave owns K 16..31
  const unsigned short* bp0 = hsrc + (n0 + (lane & 15)) * H + klo;
  const unsigned short* bp1 = bp0 + 16 * H;
  const unsigned short* ag  = Wp + (size_t)((0 * MT + mt) * KB) * FRAG + lane * 16;
  const unsigned short* ai  = Wp + (size_t)((1 * MT + mt) * KB) * FRAG + lane * 16;
  const unsigned short* af  = Wp + (size_t)((2 * MT + mt) * KB) * FRAG + lane * 16;
  const unsigned short* ao  = Wp + (size_t)((3 * MT + mt) * KB) * FRAG + lane * 16;

  for (int kb = 0; kb < KB; ++kb) {
    v16bf b0 = *(const v16bf*)(bp0 + kb * 32);
    v16bf b1 = *(const v16bf*)(bp1 + kb * 32);
    v16bf a;
    a   = *(const v16bf*)(ag + kb * FRAG);
    aG0 = __builtin_amdgcn_wmma_f32_16x16x32_bf16(false, a, false, b0, (short)0, aG0, false, false);
    aG1 = __builtin_amdgcn_wmma_f32_16x16x32_bf16(false, a, false, b1, (short)0, aG1, false, false);
    a   = *(const v16bf*)(ai + kb * FRAG);
    aI0 = __builtin_amdgcn_wmma_f32_16x16x32_bf16(false, a, false, b0, (short)0, aI0, false, false);
    aI1 = __builtin_amdgcn_wmma_f32_16x16x32_bf16(false, a, false, b1, (short)0, aI1, false, false);
    a   = *(const v16bf*)(af + kb * FRAG);
    aF0 = __builtin_amdgcn_wmma_f32_16x16x32_bf16(false, a, false, b0, (short)0, aF0, false, false);
    aF1 = __builtin_amdgcn_wmma_f32_16x16x32_bf16(false, a, false, b1, (short)0, aF1, false, false);
    a   = *(const v16bf*)(ao + kb * FRAG);
    aO0 = __builtin_amdgcn_wmma_f32_16x16x32_bf16(false, a, false, b0, (short)0, aO0, false, false);
    aO1 = __builtin_amdgcn_wmma_f32_16x16x32_bf16(false, a, false, b1, (short)0, aO1, false, false);
  }

  // Epilogue: C/D layout -> accum elem e, lane L maps to
  // row = m0 + e + (L>=16 ? 8 : 0), col = n0 + sub*16 + (L & 15).
  int hi8  = klo >> 1;            // 0 or 8
  int lcol = lane & 15;
#pragma unroll
  for (int sub = 0; sub < 2; ++sub) {
    int col = n0 + sub * 16 + lcol;
    float xv = x[col * T + t];
    unsigned short hv[8];
#pragma unroll
    for (int e = 0; e < 8; ++e) {
      int row = m0 + hi8 + e;
      float pg = (sub ? aG1 : aG0)[e] + Wgx[row] * xv + bgp[row];
      float pi = (sub ? aI1 : aI0)[e] + Wix[row] * xv + bip[row];
      float pf = (sub ? aF1 : aF0)[e] + Wfx[row] * xv + bfp[row];
      float po = (sub ? aO1 : aO0)[e] + Wox[row] * xv + bop[row];
      float gg = tanh_fast(pg);
      float ii = sigm(pi);
      float ff = sigm(pf);
      float oo = sigm(po);
      float cold = cst[row * Bn + col];
      float cn = gg * ii + cold * ff;
      cst[row * Bn + col] = cn;
      float hn = tanh_fast(cn) * oo;
      hf32[row * Bn + col] = hn;
      hv[e] = f2bf(hn);
    }
    uint4 pk;
    pk.x = (unsigned)hv[0] | ((unsigned)hv[1] << 16);
    pk.y = (unsigned)hv[2] | ((unsigned)hv[3] << 16);
    pk.z = (unsigned)hv[4] | ((unsigned)hv[5] << 16);
    pk.w = (unsigned)hv[6] | ((unsigned)hv[7] << 16);
    *(uint4*)(hdst + col * H + m0 + hi8) = pk;   // 8 contiguous bf16, 16B aligned
  }
}

// out[b][c] = W_ph @ h + b_p  (tiny: 2560 dots of length 1024, fp32)
__global__ void proj(const float* __restrict__ hf32, const float* __restrict__ Wph,
                     const float* __restrict__ bpv, float* __restrict__ out) {
  int idx = blockIdx.x * blockDim.x + threadIdx.x;
  if (idx >= Bn * Cc) return;
  int b  = idx / Cc;
  int cl = idx % Cc;
  float s = bpv[cl];
  for (int k = 0; k < H; ++k)
    s += Wph[cl * H + k] * hf32[k * Bn + b];
  out[b * Cc + cl] = s;
}

extern "C" void kernel_launch(void* const* d_in, const int* in_sizes, int n_in,
                              void* d_out, int out_size, void* d_ws, size_t ws_size,
                              hipStream_t stream) {
  const float* x   = (const float*)d_in[0];
  const float* Wgx = (const float*)d_in[1];
  const float* Wgh = (const float*)d_in[2];
  const float* Wix = (const float*)d_in[3];
  const float* Wih = (const float*)d_in[4];
  const float* Wfx = (const float*)d_in[5];
  const float* Wfh = (const float*)d_in[6];
  const float* Wox = (const float*)d_in[7];
  const float* Woh = (const float*)d_in[8];
  const float* Wph = (const float*)d_in[9];
  const float* bg  = (const float*)d_in[10];
  const float* bi  = (const float*)d_in[11];
  const float* bf  = (const float*)d_in[12];
  const float* bo  = (const float*)d_in[13];
  const float* bp  = (const float*)d_in[14];

  char* ws = (char*)d_ws;
  unsigned short* Wp   = (unsigned short*)(ws);                              // 8 MB packed bf16 weights
  unsigned short* hbf0 = (unsigned short*)(ws + (8u << 20));                 // 512 KB
  unsigned short* hbf1 = (unsigned short*)(ws + (8u << 20) + (512u << 10));  // 512 KB
  float* cst  = (float*)(ws + (9u  << 20));                                  // 1 MB cell state
  float* hf32 = (float*)(ws + (10u << 20));                                  // 1 MB fp32 h for projection

  pack_w<<<(4 * H * H) / 256, 256, 0, stream>>>(Wgh, Wih, Wfh, Woh, Wp);
  init_state<<<(H * Bn) / 256, 256, 0, stream>>>(cst, hbf0);

  for (int t = 0; t < T; ++t) {
    const unsigned short* hs = (t & 1) ? hbf1 : hbf0;
    unsigned short*       hd = (t & 1) ? hbf0 : hbf1;
    lstm_step<<<64, 256, 0, stream>>>(Wp, hs, hd, hf32, cst, x,
                                      Wgx, Wix, Wfx, Wox, bg, bi, bf, bo, t);
  }

  proj<<<(Bn * Cc + 255) / 256, 256, 0, stream>>>(hf32, Wph, bp, (float*)d_out);
}
